// SinkhornAttention_13348758355987
// MI455X (gfx1250) — matrix operand order repaired and checked by reading, not compiled, not run on hardware
//
#include <hip/hip_runtime.h>
#include <math.h>

// ---------------------------------------------------------------------------
// SinkhornAttention on MI455X (gfx1250, wave32, WMMA + async global->LDS)
// d_out = [ out (4096*8*1024) | attn_weights (8*256*256) | log_alpha (8*256*256) ]
// All GEMM operands staged once as bf16. Big GEMMs run a double-buffered
// software pipeline: global_load_async_to_lds_b64 (ASYNCcnt) for tile t+1
// overlapped with v_wmma_f32_16x16x32_bf16 on tile t.
// ---------------------------------------------------------------------------

typedef __attribute__((ext_vector_type(16))) __bf16 v16bf;
typedef __attribute__((ext_vector_type(8)))  float  v8f;

#define TILE_M 128
#define TILE_N 128
#define TILE_K 32
#define LDST   40   // LDS row stride in ushorts (mult of 8 -> 16B-aligned b128 frags)

__device__ __forceinline__ unsigned int f2bf(float f) {
  unsigned int u = __float_as_uint(f);
  u += 0x7FFFu + ((u >> 16) & 1u);      // round-to-nearest-even
  return u >> 16;
}

union FragAB { v16bf v; uint4 u[2]; };

// async copy of one 8-byte chunk per lane: global (saddr + voff) -> LDS[laddr]
__device__ __forceinline__ void async_b64(unsigned laddr, unsigned voff, const void* sbase) {
  asm volatile("global_load_async_to_lds_b64 %0, %1, %2"
               :: "v"(laddr), "v"(voff), "s"(sbase) : "memory");
}
__device__ __forceinline__ void wait_async0() {
  asm volatile("s_wait_asynccnt 0" ::: "memory");
}

// C = scale * (A @ B^T) + bias, bf16 WMMA inner, f32 accumulate.
//  AMODE 0: A fp32 MxK row-major (convert into LDS)
//  AMODE 1: A bf16 MxK row-major (async DMA to LDS)
//  BMODE 1: B bf16 NxK row-major (async DMA to LDS)
//  BMODE 2: B = bf16 vb-view of value (S,B,E): B[n][k]=value[k*16+n/1024][z][n%1024]
//  CMODE 0: C MxN row-major (+z*sCz)   CMODE 1: attn scatter to bf16 (T,B,E)
//  COUT  0: C fp32                     COUT  1: C bf16
template<int AMODE, int BMODE, int CMODE, int COUT>
__global__ __launch_bounds__(256)
void gemm_wmma(const void* __restrict__ A,
               const void* __restrict__ Bm,
               void* __restrict__ C,
               const float* __restrict__ bias,
               int N, int K,
               long long sAz, long long sBz, long long sCz,
               float scale)
{
  constexpr bool PIPE = (AMODE == 1 && BMODE == 1);   // fully-async operands
  constexpr int  NBUF = PIPE ? 2 : 1;

  __shared__ unsigned short lA[NBUF][TILE_M][LDST];
  __shared__ unsigned short lB[NBUF][TILE_N][LDST];

  const int tid  = threadIdx.x;
  const int wave = tid >> 5;
  const int lane = tid & 31;
  const int z    = blockIdx.z;
  const int n0   = blockIdx.x * TILE_N;
  const int m0   = blockIdx.y * TILE_M;
  const int lh   = lane >> 4;
  const int ln   = lane & 15;

  v8f acc[8];
#pragma unroll
  for (int j = 0; j < 8; ++j)
#pragma unroll
    for (int i = 0; i < 8; ++i) acc[j][i] = 0.0f;

  auto loadA = [&](int k0, int buf) {
    if constexpr (AMODE == 1) {          // bf16: pure async DMA
      const unsigned short* base =
          (const unsigned short*)A + (size_t)z * sAz + (size_t)m0 * K + k0;
      unsigned l0 = (unsigned)(size_t)&lA[buf][0][0];
#pragma unroll
      for (int i = 0; i < 4; ++i) {
        int chunk = (wave << 7) + (i << 5) + lane;   // 0..1023 (8B chunks)
        int r = chunk >> 3, c8 = chunk & 7;
        async_b64(l0 + (unsigned)(r * (LDST * 2) + c8 * 8),
                  (unsigned)(r * K * 2 + c8 * 8), base);
      }
    } else {                             // fp32: convert into LDS
      int r  = tid >> 1;
      int kc = (tid & 1) << 4;
      const float4* src = (const float4*)((const float*)A + (size_t)z * sAz +
                                          (size_t)(m0 + r) * K + (k0 + kc));
      float4 f0 = src[0], f1 = src[1], f2 = src[2], f3 = src[3];
      unsigned int* d = (unsigned int*)&lA[buf][r][kc];
      d[0]=f2bf(f0.x)|(f2bf(f0.y)<<16); d[1]=f2bf(f0.z)|(f2bf(f0.w)<<16);
      d[2]=f2bf(f1.x)|(f2bf(f1.y)<<16); d[3]=f2bf(f1.z)|(f2bf(f1.w)<<16);
      d[4]=f2bf(f2.x)|(f2bf(f2.y)<<16); d[5]=f2bf(f2.z)|(f2bf(f2.w)<<16);
      d[6]=f2bf(f3.x)|(f2bf(f3.y)<<16); d[7]=f2bf(f3.z)|(f2bf(f3.w)<<16);
    }
  };

  auto loadB = [&](int k0, int buf) {
    if constexpr (BMODE == 1) {          // bf16 NxK: pure async DMA
      const unsigned short* base =
          (const unsigned short*)Bm + (size_t)z * sBz + (size_t)n0 * K + k0;
      unsigned l0 = (unsigned)(size_t)&lB[buf][0][0];
#pragma unroll
      for (int i = 0; i < 4; ++i) {
        int chunk = (wave << 7) + (i << 5) + lane;
        int r = chunk >> 3, c8 = chunk & 7;
        async_b64(l0 + (unsigned)(r * (LDST * 2) + c8 * 8),
                  (unsigned)(r * K * 2 + c8 * 8), base);
      }
    } else {                             // vb-view of bf16 value (S,B,E)
      int kk = tid >> 3;                 // 0..31
      int nn = (tid & 7) << 4;           // 0..112
      int n  = n0 + nn;
      int jj = n >> 10;
      int e  = n & 1023;
      const unsigned short* src = (const unsigned short*)Bm +
          (((size_t)((k0 + kk) * 16 + jj)) * 8 + z) * 1024 + e;
      union { uint4 q[2]; unsigned short s[16]; } u;
      u.q[0] = ((const uint4*)src)[0];
      u.q[1] = ((const uint4*)src)[1];
#pragma unroll
      for (int t = 0; t < 16; ++t) lB[buf][nn + t][kk] = u.s[t];
    }
  };

  auto compute = [&](int buf) {
    FragAB a;
    int r = (wave << 4) + ln;
    a.u[0] = *(const uint4*)&lA[buf][r][lh << 3];
    a.u[1] = *(const uint4*)&lA[buf][r][16 + (lh << 3)];
#pragma unroll
    for (int j = 0; j < 8; ++j) {
      FragAB b;
      int c = (j << 4) + ln;
      b.u[0] = *(const uint4*)&lB[buf][c][lh << 3];
      b.u[1] = *(const uint4*)&lB[buf][c][16 + (lh << 3)];
      acc[j] = __builtin_amdgcn_wmma_f32_16x16x32_bf16(
          false, a.v, false, b.v, (short)0, acc[j], false, false);
    }
  };

  if constexpr (PIPE) {
    // double-buffered: DMA tile t+1 while WMMA consumes tile t
    loadA(0, 0); loadB(0, 0);
    wait_async0();
    __syncthreads();
    const int nk = K / TILE_K;
    for (int t = 0; t < nk; ++t) {
      int cur = t & 1;
      if (t + 1 < nk) { loadA((t + 1) * TILE_K, cur ^ 1); loadB((t + 1) * TILE_K, cur ^ 1); }
      compute(cur);
      if (t + 1 < nk) wait_async0();
      __syncthreads();
    }
  } else {
    for (int k0 = 0; k0 < K; k0 += TILE_K) {
      __syncthreads();
      loadA(k0, 0);
      loadB(k0, 0);
      __syncthreads();
      compute(0);
    }
  }

  // ---- epilogue ----
#pragma unroll
  for (int j = 0; j < 8; ++j) {
    int n = n0 + (j << 4) + ln;
    float bv = (bias != nullptr) ? bias[n] : 0.0f;
#pragma unroll
    for (int i = 0; i < 8; ++i) {
      int m = m0 + (wave << 4) + (lh << 3) + i;
      float val = acc[j][i] * scale + bv;
      if constexpr (CMODE == 0) {
        size_t off = (size_t)z * sCz + (size_t)m * N + n;
        if constexpr (COUT == 0) ((float*)C)[off] = val;
        else ((unsigned short*)C)[off] = (unsigned short)f2bf(val);
      } else {   // attn scatter -> bf16 (T,B,E)
        int t = m * 16 + (n >> 10);
        ((unsigned short*)C)[((size_t)t * 8 + z) * 1024 + (n & 1023)] =
            (unsigned short)f2bf(val);
      }
    }
  }
}

// mean over 16-row buckets of (T,B,E) -> bf16 (B, T/16, E), (b*256+i)*1024+e
__global__ void pool16(const float* __restrict__ src, unsigned short* __restrict__ dst,
                       int total) {
  int idx = blockIdx.x * blockDim.x + threadIdx.x;
  if (idx >= total) return;
  int e = idx & 1023;
  int r = idx >> 10;
  int i = r & 255;
  int b = r >> 8;
  const float* p = src + ((size_t)(i * 16) * 8 + b) * 1024 + e;
  float s = 0.0f;
#pragma unroll
  for (int j = 0; j < 16; ++j) s += p[(size_t)j * 8192];
  dst[idx] = (unsigned short)f2bf(s * 0.0625f);
}

__global__ void cvt_bf16(const float* __restrict__ src, unsigned short* __restrict__ dst,
                         int total) {
  int idx = blockIdx.x * blockDim.x + threadIdx.x;
  if (idx >= total) return;
  dst[idx] = (unsigned short)f2bf(src[idx]);
}

// la = (log_alpha + gumbel) / tau  (deterministic hash-based noise)
__global__ void sink_init(const float* __restrict__ loga, float* __restrict__ la, int total) {
  int idx = blockIdx.x * blockDim.x + threadIdx.x;
  if (idx >= total) return;
  unsigned long long h = (unsigned long long)idx * 0x9E3779B97F4A7C15ull + 42ull;
  h ^= h >> 30; h *= 0xBF58476D1CE4E5B9ull;
  h ^= h >> 27; h *= 0x94D049BB133111EBull;
  h ^= h >> 31;
  float u = (float)(h >> 40) * (1.0f / 16777216.0f);
  float g = -logf(-logf(u + 1e-6f) + 1e-6f);
  la[idx] = (loga[idx] + g) * (1.0f / 0.75f);
}

// subtract logsumexp over last dim; wave per row (256 elems, 8/lane)
__global__ __launch_bounds__(256) void sink_rownorm(float* __restrict__ la) {
  int wave = threadIdx.x >> 5, lane = threadIdx.x & 31;
  int row = blockIdx.x * 8 + wave;
  float* p = la + (size_t)row * 256 + lane * 8;
  float x[8];
#pragma unroll
  for (int t = 0; t < 8; ++t) x[t] = p[t];
  float m = x[0];
#pragma unroll
  for (int t = 1; t < 8; ++t) m = fmaxf(m, x[t]);
  for (int off = 16; off > 0; off >>= 1) m = fmaxf(m, __shfl_xor(m, off, 32));
  float s = 0.0f;
#pragma unroll
  for (int t = 0; t < 8; ++t) s += expf(x[t] - m);
  for (int off = 16; off > 0; off >>= 1) s += __shfl_xor(s, off, 32);
  float lse = m + logf(s);
#pragma unroll
  for (int t = 0; t < 8; ++t) p[t] = x[t] - lse;
}

// subtract logsumexp over rows; thread per column, one block per batch
__global__ __launch_bounds__(256) void sink_colnorm(float* __restrict__ la) {
  int b = blockIdx.x, j = threadIdx.x;
  float* base = la + (size_t)b * 65536 + j;
  float m = -1e30f, s = 0.0f;
  for (int i = 0; i < 256; ++i) {
    float x = base[(size_t)i * 256];
    float nm = fmaxf(m, x);
    s = s * expf(m - nm) + expf(x - nm);
    m = nm;
  }
  float lse = m + logf(s);
  for (int i = 0; i < 256; ++i) base[(size_t)i * 256] -= lse;
}

__global__ void sink_exp(const float* __restrict__ la, float* __restrict__ aw, int total) {
  int idx = blockIdx.x * blockDim.x + threadIdx.x;
  if (idx >= total) return;
  aw[idx] = expf(la[idx]);
}

extern "C" void kernel_launch(void* const* d_in, const int* in_sizes, int n_in,
                              void* d_out, int out_size, void* d_ws, size_t ws_size,
                              hipStream_t stream) {
  (void)in_sizes; (void)n_in; (void)out_size; (void)ws_size;

  const float* query = (const float*)d_in[0];  // (4096, 8, 1024)
  const float* key   = (const float*)d_in[1];
  const float* value = (const float*)d_in[2];
  const float* wq    = (const float*)d_in[3];  // (1024, 1024)
  const float* bq    = (const float*)d_in[4];
  const float* wk    = (const float*)d_in[5];
  const float* bk    = (const float*)d_in[6];
  const float* wo    = (const float*)d_in[7];
  const float* bo    = (const float*)d_in[8];

  float* out  = (float*)d_out;          // 33554432
  float* aw   = out + 33554432;         // 524288
  float* loga = aw + 524288;            // 524288

  // workspace (bf16 staging + sinkhorn scratch)
  unsigned short* value_bf = (unsigned short*)d_ws;        // 33554432
  unsigned short* attn_bf  = value_bf + 33554432;          // 33554432
  unsigned short* qpool_bf = attn_bf  + 33554432;          // 2097152
  unsigned short* kpool_bf = qpool_bf + 2097152;
  unsigned short* qb_bf    = kpool_bf + 2097152;
  unsigned short* kb_bf    = qb_bf    + 2097152;
  unsigned short* wq_bf    = kb_bf    + 2097152;           // 1048576
  unsigned short* wk_bf    = wq_bf    + 1048576;
  unsigned short* wo_bf    = wk_bf    + 1048576;
  float*          la       = (float*)(wo_bf + 1048576);    // 524288 f32

  // 0) stage weights + value as bf16 (once)
  cvt_bf16<<<4096,   256, 0, stream>>>(wq, wq_bf, 1048576);
  cvt_bf16<<<4096,   256, 0, stream>>>(wk, wk_bf, 1048576);
  cvt_bf16<<<4096,   256, 0, stream>>>(wo, wo_bf, 1048576);
  cvt_bf16<<<131072, 256, 0, stream>>>(value, value_bf, 33554432);

  // 1) bucket-mean pooling (projection commutes with mean) -> bf16
  pool16<<<8192, 256, 0, stream>>>(query, qpool_bf, 2097152);
  pool16<<<8192, 256, 0, stream>>>(key,   kpool_bf, 2097152);

  // 2) pooled projections (2048x1024x1024), pipelined async tiles, bf16 out
  gemm_wmma<1,1,0,1><<<dim3(8, 16, 1), 256, 0, stream>>>(
      qpool_bf, wq_bf, qb_bf, bq, 1024, 1024, 0, 0, 0, 1.0f);
  gemm_wmma<1,1,0,1><<<dim3(8, 16, 1), 256, 0, stream>>>(
      kpool_bf, wk_bf, kb_bf, bk, 1024, 1024, 0, 0, 0, 1.0f);

  // 3) logits = SCALE * qb @ kb^T -> log_alpha (fp32 out)
  gemm_wmma<1,1,0,0><<<dim3(2, 2, 8), 256, 0, stream>>>(
      qb_bf, kb_bf, loga, nullptr, 256, 1024,
      262144, 262144, 65536, 0.03125f);

  // 4) gumbel-sinkhorn (8 x row/col log-normalization)
  sink_init<<<2048, 256, 0, stream>>>(loga, la, 524288);
  for (int it = 0; it < 8; ++it) {
    sink_rownorm<<<256, 256, 0, stream>>>(la);
    sink_colnorm<<<8, 256, 0, stream>>>(la);
  }
  sink_exp<<<2048, 256, 0, stream>>>(la, aw, 524288);

  // 5) attend: attn = aw @ vb, scattered to bf16 (T,B,E)
  gemm_wmma<0,2,1,1><<<dim3(128, 2, 8), 256, 0, stream>>>(
      aw, value_bf, attn_bf, nullptr, 16384, 256,
      65536, 0, 0, 1.0f);

  // 6) output projection: out = attn @ wo^T + bo (32768x1024x1024), pipelined async
  gemm_wmma<1,1,0,0><<<dim3(8, 256, 1), 256, 0, stream>>>(
      attn_bf, wo_bf, out, bo, 1024, 1024, 0, 0, 0, 1.0f);
}
